// QuantumFidelityLoss_80367428042774
// MI455X (gfx1250) — matrix operand order repaired and checked by reading, not compiled, not run on hardware
//
#include <hip/hip_runtime.h>
#include <hip/hip_bf16.h>

#define BATCH 8192
#define DIM   2048

typedef __attribute__((ext_vector_type(8)))  float  v8f;
typedef __attribute__((ext_vector_type(8)))  __bf16 v8bf;
typedef __attribute__((ext_vector_type(16))) __bf16 v16bf;

// ---------------------------------------------------------------------------
// round-to-nearest-even f32 -> bf16 (bit manipulation; no NaNs in this data)
// ---------------------------------------------------------------------------
__device__ __forceinline__ unsigned short f32_to_bf16(float f) {
  unsigned int u = __float_as_uint(f);
  unsigned int rounding = 0x7FFFu + ((u >> 16) & 1u);
  return (unsigned short)((u + rounding) >> 16);
}

// ---------------------------------------------------------------------------
// Stage: convert f32 [BATCH, DIM] -> bf16 transposed [DIM, BATCH] via LDS tile
// so the Gram kernel gets K contiguous in memory (wide b128 fragment loads).
// blockIdx.z selects latent (0) vs target (1).
// ---------------------------------------------------------------------------
__global__ __launch_bounds__(256) void cvt_transpose_kernel(
    const float* __restrict__ inL, const float* __restrict__ inT,
    unsigned short* __restrict__ outL, unsigned short* __restrict__ outT) {
  const float* __restrict__ in        = blockIdx.z ? inT : inL;
  unsigned short* __restrict__ out    = blockIdx.z ? outT : outL;
  const int kBase = blockIdx.x * 64;   // batch-row tile
  const int cBase = blockIdx.y * 64;   // column tile

  __shared__ unsigned short tile[64][72];  // [c][k], 144B rows (8B aligned)

  // load phase: coalesced along columns; store transposed into LDS
#pragma unroll
  for (int i = 0; i < 16; ++i) {
    const int idx = threadIdx.x + i * 256;
    const int r = idx >> 6;   // k offset 0..63
    const int c = idx & 63;   // col offset 0..63
    tile[c][r] = f32_to_bf16(in[(size_t)(kBase + r) * DIM + (size_t)(cBase + c)]);
  }
  __syncthreads();

  // store phase: 8-byte chunks, contiguous in k (coalesced in the output)
#pragma unroll
  for (int v = 0; v < 4; ++v) {
    const int id = threadIdx.x + v * 256;
    const int c = id >> 4;    // 0..63
    const int q = id & 15;    // k-quad 0..15
    const unsigned long long val = *(const unsigned long long*)&tile[c][q * 4];
    *(unsigned long long*)&out[(size_t)(cBase + c) * BATCH + (size_t)(kBase + q * 4)] = val;
  }
}

// ---------------------------------------------------------------------------
// bf16 WMMA fragment loads from transposed bf16 array p [DIM, BATCH].
// A-matrix 16x32: lane l holds col (l&15), k = base+(l>>4)*16 +
//   {0..7} in dwords 0-3 and +8..+15 in dwords 4-7  -> two 16B loads.
// B-matrix 32x16: lane l holds col (l&15), 16 consecutive k starting at
//   base+(l>>4)*16 -> one contiguous 32B access.
// ---------------------------------------------------------------------------
__device__ __forceinline__ v16bf load_a_frag(const __bf16* __restrict__ p,
                                             int colBase, int kc, int lane) {
  const size_t base = (size_t)(colBase + (lane & 15)) * BATCH
                    + (size_t)(kc * 32 + ((lane >> 4) << 4));
  v8bf lo = *(const v8bf*)(p + base);        // k .. k+7
  v8bf hi = *(const v8bf*)(p + base + 8);    // k+8 .. k+15
  return __builtin_shufflevector(lo, hi, 0, 1, 2, 3, 4, 5, 6, 7,
                                         8, 9, 10, 11, 12, 13, 14, 15);
}

__device__ __forceinline__ v16bf load_b_frag(const __bf16* __restrict__ p,
                                             int colBase, int kc, int lane) {
  const size_t base = (size_t)(colBase + (lane & 15)) * BATCH
                    + (size_t)(kc * 32 + ((lane >> 4) << 4));
  return *(const v16bf*)(p + base);          // k .. k+15 contiguous
}

// ---------------------------------------------------------------------------
// Fused Gram-trace: each wave accumulates a 32x32 tile of BOTH rho and sigma
// with v_wmma_f32_16x16x32_bf16, multiplies them elementwise in registers,
// and reduces into a single global scalar. Gram matrices never hit memory.
// Block = 8 waves (2x4) -> 64x128 block tile; grid (16,32).
// ---------------------------------------------------------------------------
__global__ __launch_bounds__(256) void gram_trace_kernel(
    const __bf16* __restrict__ Lt,   // [DIM, BATCH] bf16 (latent, transposed)
    const __bf16* __restrict__ Tt,   // [DIM, BATCH] bf16 (target, transposed)
    float* __restrict__ acc_out) {
  const int lane = threadIdx.x & 31;
  const int w    = threadIdx.x >> 5;        // 0..7
  const int wm   = w >> 2;                  // 0..1
  const int wn   = w & 3;                   // 0..3
  const int iBase = blockIdx.y * 64  + wm * 32;   // Gram row-block
  const int jBase = blockIdx.x * 128 + wn * 32;   // Gram col-block

  v8f accR[2][2];  // rho tiles   (from L)
  v8f accS[2][2];  // sigma tiles (from T)
#pragma unroll
  for (int mi = 0; mi < 2; ++mi)
#pragma unroll
    for (int nj = 0; nj < 2; ++nj) {
      accR[mi][nj] = (v8f)0.0f;
      accS[mi][nj] = (v8f)0.0f;
    }

  for (int kc = 0; kc < BATCH / 32; ++kc) {
    v16bf aL[2], aT[2], bL[2], bT[2];
#pragma unroll
    for (int mi = 0; mi < 2; ++mi) {
      aL[mi] = load_a_frag(Lt, iBase + mi * 16, kc, lane);
      aT[mi] = load_a_frag(Tt, iBase + mi * 16, kc, lane);
    }
#pragma unroll
    for (int nj = 0; nj < 2; ++nj) {
      bL[nj] = load_b_frag(Lt, jBase + nj * 16, kc, lane);
      bT[nj] = load_b_frag(Tt, jBase + nj * 16, kc, lane);
    }
#pragma unroll
    for (int mi = 0; mi < 2; ++mi)
#pragma unroll
      for (int nj = 0; nj < 2; ++nj) {
        accR[mi][nj] = __builtin_amdgcn_wmma_f32_16x16x32_bf16(
            false, aL[mi], false, bL[nj], (short)0, accR[mi][nj], false, false);
        accS[mi][nj] = __builtin_amdgcn_wmma_f32_16x16x32_bf16(
            false, aT[mi], false, bT[nj], (short)0, accS[mi][nj], false, false);
      }
  }

  // trace contribution: elementwise rho_tile * sigma_tile (same VGPR layout)
  float s = 0.0f;
#pragma unroll
  for (int mi = 0; mi < 2; ++mi)
#pragma unroll
    for (int nj = 0; nj < 2; ++nj)
#pragma unroll
      for (int v = 0; v < 8; ++v)
        s += accR[mi][nj][v] * accS[mi][nj][v];

  // wave32 reduction
#pragma unroll
  for (int off = 16; off > 0; off >>= 1)
    s += __shfl_down(s, off, 32);

  __shared__ float red[8];
  if (lane == 0) red[w] = s;
  __syncthreads();
  if (threadIdx.x == 0) {
    float t = 0.0f;
#pragma unroll
    for (int i = 0; i < 8; ++i) t += red[i];
    atomicAdd(acc_out, t);
  }
}

__global__ void init_kernel(float* ws) { ws[0] = 0.0f; }

__global__ void finalize_kernel(const float* __restrict__ ws,
                                float* __restrict__ out) {
  const float invB2 = 1.0f / ((float)BATCH * (float)BATCH);
  float fid = ws[0] * invB2;
  fid = fminf(fmaxf(fid, 0.0f), 1.0f);
  out[0] = 0.1f * fabsf(fid - 0.95f);
}

extern "C" void kernel_launch(void* const* d_in, const int* in_sizes, int n_in,
                              void* d_out, int out_size, void* d_ws, size_t ws_size,
                              hipStream_t stream) {
  const float* L = (const float*)d_in[0];
  const float* T = (const float*)d_in[1];
  float* out = (float*)d_out;

  // workspace layout: [0,4) scalar accumulator | 256: Lt (32MB) | +32MB: Tt
  char* ws_base = (char*)d_ws;
  float* acc = (float*)ws_base;
  unsigned short* Lt = (unsigned short*)(ws_base + 256);
  unsigned short* Tt = (unsigned short*)(ws_base + 256 + (size_t)DIM * BATCH * 2);

  init_kernel<<<dim3(1), dim3(1), 0, stream>>>(acc);

  dim3 cvt_grid(BATCH / 64, DIM / 64, 2);   // (128, 32, 2)
  cvt_transpose_kernel<<<cvt_grid, dim3(256), 0, stream>>>(L, T, Lt, Tt);

  dim3 grid(DIM / 128, DIM / 64);           // (16, 32) = 512 workgroups
  gram_trace_kernel<<<grid, dim3(256), 0, stream>>>(
      (const __bf16*)Lt, (const __bf16*)Tt, acc);

  finalize_kernel<<<dim3(1), dim3(1), 0, stream>>>(acc, out);
}
